// BoundaryPredictor1_55551107006573
// MI455X (gfx1250) — matrix operand order repaired and checked by reading, not compiled, not run on hardware
//
#include <hip/hip_runtime.h>
#include <math.h>

// ---------------- problem constants ----------------
constexpr int BS = 4, S = 2048, D = 1024, H = 4096;
constexpr int TOK = BS * S;                 // 8192 tokens
constexpr float TEMP = 1.0f, THRESH = 0.5f, PRIOR = 0.2f, EPS = 1e-9f;

// GEMM tiling
constexpr int BM = 64, BN = 64, BK = 32;
constexpr int MBLK = TOK / BM;              // 128
constexpr int NBLK = H / BN;                // 64
constexpr int NK   = D / BK;                // 32 k-steps
constexpr int LDP  = BK + 8;                // padded LDS row: 40 bf16 = 80 B

// output layout (floats)
constexpr size_t POOLED_SZ = (size_t)BS * S * D;     // 8,388,608
constexpr size_t LOSS_OFF  = POOLED_SZ;
constexpr size_t NUMB_OFF  = POOLED_SZ + 1;
constexpr size_t TOT_OFF   = POOLED_SZ + 2;
constexpr size_t SMASK_OFF = POOLED_SZ + 3;

// ---------------- WMMA types ----------------
typedef __bf16 v16bf __attribute__((ext_vector_type(16)));
typedef float  v8f   __attribute__((ext_vector_type(8)));

union V16 { v16bf v; uint4 q[2]; };          // fragment = 2x ds_load_b128
union H8  { __bf16 b[8]; uint4 q; };         // 8 bf16 = one b128 store

// ============================================================
// Kernel 1: fused gate GEMM: partial[nblk][tok] = sum over this
// n-block's 64 columns of relu(hidden@W1 + b1) * W2
// ============================================================
__global__ __launch_bounds__(256)
void gate_gemm_kernel(const float* __restrict__ hidden,
                      const float* __restrict__ W1,
                      const float* __restrict__ b1,
                      const float* __restrict__ W2,
                      float* __restrict__ partial /* [NBLK][TOK] */) {
    __shared__ __bf16 Ah[BM][LDP];   // hi plane, token-major   (5 KB)
    __shared__ __bf16 Al[BM][LDP];   // lo plane
    __shared__ __bf16 Bh[BN][LDP];   // hi plane, n-major
    __shared__ __bf16 Bl[BN][LDP];   // lo plane
    __shared__ float  epi[4][16];    // cross-wave-group combine

    const int tid  = threadIdx.x;
    const int lane = tid & 31;
    const int wave = tid >> 5;             // 0..7
    const int mt   = wave & 3;             // wave's M sub-tile (16 rows)
    const int ng   = wave >> 2;            // wave's N half (2 sub-tiles each)

    const int m0 = blockIdx.y * BM;        // token base
    const int n0 = blockIdx.x * BN;        // hidden-unit base

    v8f c[2];
    c[0] = (v8f){0.f,0.f,0.f,0.f,0.f,0.f,0.f,0.f};
    c[1] = (v8f){0.f,0.f,0.f,0.f,0.f,0.f,0.f,0.f};

    // staging coordinates
    const int ar  = tid >> 2;              // 0..63 token row (A)
    const int ac  = (tid & 3) * 8;         // k seg 0/8/16/24 (A)
    const int bnl = tid & 63;              // 0..63 n col (B)
    const int bks = (tid >> 6) * 8;        // k seg 0/8/16/24 (B)

    const int lo16  = lane & 15;
    const int khalf = lane >> 4;

    // ---- software pipeline: preload k-panel 0 ----
    float4 a0, a1;            // A: 8 consecutive f32
    float  bv[8];             // B: 8 k-consecutive f32 at fixed n
    {
        const float4* p = (const float4*)&hidden[(size_t)(m0 + ar) * D + ac];
        a0 = p[0]; a1 = p[1];
        #pragma unroll
        for (int j = 0; j < 8; ++j)
            bv[j] = W1[(size_t)(bks + j) * H + n0 + bnl];
    }

    for (int kk = 0; kk < NK; ++kk) {
        // ---- convert current panel to bf16 hi/lo planes in LDS ----
        {
            float xa[8] = {a0.x,a0.y,a0.z,a0.w, a1.x,a1.y,a1.z,a1.w};
            H8 hA, lA, hB, lB;
            #pragma unroll
            for (int j = 0; j < 8; ++j) {
                __bf16 hh = (__bf16)xa[j];
                hA.b[j] = hh;
                lA.b[j] = (__bf16)(xa[j] - (float)hh);
            }
            #pragma unroll
            for (int j = 0; j < 8; ++j) {
                __bf16 hh = (__bf16)bv[j];
                hB.b[j] = hh;
                lB.b[j] = (__bf16)(bv[j] - (float)hh);
            }
            *(uint4*)&Ah[ar][ac]   = hA.q;
            *(uint4*)&Al[ar][ac]   = lA.q;
            *(uint4*)&Bh[bnl][bks] = hB.q;
            *(uint4*)&Bl[bnl][bks] = lB.q;
        }
        __syncthreads();

        // ---- issue next panel's global loads (overlap with WMMA) ----
        if (kk + 1 < NK) {
            const int k0n = (kk + 1) * BK;
            const float4* p = (const float4*)&hidden[(size_t)(m0 + ar) * D + k0n + ac];
            a0 = p[0]; a1 = p[1];
            #pragma unroll
            for (int j = 0; j < 8; ++j)
                bv[j] = W1[(size_t)(k0n + bks + j) * H + n0 + bnl];
        }

        // ---- A fragments (ISA 16-bit A 16x32 layout): contiguous b128s ----
        V16 aH, aL;
        {
            const __bf16* rh = &Ah[mt * 16 + lo16][0];
            const __bf16* rl = &Al[mt * 16 + lo16][0];
            aH.q[0] = *(const uint4*)(rh + khalf * 8);
            aH.q[1] = *(const uint4*)(rh + 16 + khalf * 8);
            aL.q[0] = *(const uint4*)(rl + khalf * 8);
            aL.q[1] = *(const uint4*)(rl + 16 + khalf * 8);
        }
        // ---- B fragments + 3x WMMA per N sub-tile (bf16x3) ----
        #pragma unroll
        for (int t = 0; t < 2; ++t) {
            const int nt = ng * 2 + t;
            V16 bH, bL;
            const __bf16* rh = &Bh[nt * 16 + lo16][khalf * 8];
            const __bf16* rl = &Bl[nt * 16 + lo16][khalf * 8];
            bH.q[0] = *(const uint4*)(rh);
            bH.q[1] = *(const uint4*)(rh + 16);
            bL.q[0] = *(const uint4*)(rl);
            bL.q[1] = *(const uint4*)(rl + 16);
            c[t] = __builtin_amdgcn_wmma_f32_16x16x32_bf16(false, aH.v, false, bH.v, (short)0, c[t], false, false);
            c[t] = __builtin_amdgcn_wmma_f32_16x16x32_bf16(false, aH.v, false, bL.v, (short)0, c[t], false, false);
            c[t] = __builtin_amdgcn_wmma_f32_16x16x32_bf16(false, aL.v, false, bH.v, (short)0, c[t], false, false);
        }
        __syncthreads();
    }

    // ---- epilogue: relu(c + b1) * W2, reduce over this wave's 32 columns ----
    float acc[8] = {0.f,0.f,0.f,0.f,0.f,0.f,0.f,0.f};
    #pragma unroll
    for (int t = 0; t < 2; ++t) {
        const int ncol = n0 + (ng * 2 + t) * 16 + lo16;
        const float w2 = W2[ncol];
        const float bb = b1[ncol];
        #pragma unroll
        for (int r = 0; r < 8; ++r)
            acc[r] += fmaxf(c[t][r] + bb, 0.f) * w2;
    }
    // sum across the 16 lanes of each half (columns of the tile)
    #pragma unroll
    for (int off = 1; off < 16; off <<= 1)
        #pragma unroll
        for (int r = 0; r < 8; ++r)
            acc[r] += __shfl_xor(acc[r], off, 32);

    // combine the two N wave-groups deterministically via LDS
    if (ng == 1 && (lane == 0 || lane == 16)) {
        #pragma unroll
        for (int r = 0; r < 8; ++r)
            epi[mt][khalf * 8 + r] = acc[r];
    }
    __syncthreads();
    if (ng == 0 && (lane == 0 || lane == 16)) {
        #pragma unroll
        for (int r = 0; r < 8; ++r) {
            int rowl = mt * 16 + khalf * 8 + r;
            partial[(size_t)blockIdx.x * TOK + m0 + rowl] = acc[r] + epi[mt][khalf * 8 + r];
        }
    }
}

// ============================================================
// Kernel 2: deterministic reduction of per-N-block partials
// ============================================================
__global__ __launch_bounds__(256)
void reduce_logits_kernel(const float* __restrict__ partial, float* __restrict__ logits) {
    int g = blockIdx.x * 256 + threadIdx.x;
    if (g >= TOK) return;
    float s = 0.f;
    #pragma unroll 8
    for (int i = 0; i < NBLK; ++i) s += partial[(size_t)i * TOK + g];
    logits[g] = s;
}

// ============================================================
// Kernel 3: per-row boundaries, seg bookkeeping, short_mask
// ============================================================
__global__ __launch_bounds__(256)
void boundary_kernel(const float* __restrict__ logits,
                     const float* __restrict__ mask,
                     const float* __restrict__ noise,
                     const float* __restrict__ b2,
                     int* __restrict__ seg_start,   // [BS][S]
                     int* __restrict__ seg_count,   // [BS][S]
                     float* __restrict__ row_stats, // [BS][2] = {count,total}
                     float* __restrict__ out) {
    __shared__ float hard_sh[S];
    __shared__ int   segid_sh[S];
    __shared__ int   firstPad;
    __shared__ int   cnt_sh;

    const int b = blockIdx.x;
    const int tid = threadIdx.x;
    if (tid == 0) firstPad = S;
    __syncthreads();

    const float bias2 = b2[0];
    for (int s = tid; s < S; s += 256) {
        float m = mask[b * S + s];
        if (m == 0.f) atomicMin(&firstPad, s);
        float u = noise[b * S + s];
        float logistic = logf(u) - log1pf(-u);
        float z = (logits[b * S + s] + bias2 + logistic) / TEMP;
        float soft = 1.f / (1.f + expf(-z));
        float h = (soft > THRESH) ? 1.f : 0.f;
        hard_sh[s] = h * m;
        seg_start[b * S + s] = 0;
        seg_count[b * S + s] = 0;
    }
    __syncthreads();

    if (tid == 0) {
        int fp = firstPad;
        if (fp > 0 && fp < S) hard_sh[fp - 1] = fmaxf(hard_sh[fp - 1], 1.f);
        int seg = 0;
        for (int s = 0; s < S; ++s) {          // exclusive cumsum -> seg id
            segid_sh[s] = seg;
            seg += (int)hard_sh[s];
        }
        cnt_sh = seg;
        row_stats[b * 2 + 0] = (float)seg;      // row boundary count
        row_stats[b * 2 + 1] = (float)fp;       // row valid length
    }
    __syncthreads();

    // segment starts (segments are contiguous token runs)
    for (int s = tid; s < S; s += 256) {
        int id = segid_sh[s];
        if (s == 0 || segid_sh[s - 1] != id) seg_start[b * S + id] = s;
    }
    __syncthreads();
    // segment counts
    for (int s = tid; s < S; s += 256) {
        int id = segid_sh[s];
        if (s == S - 1 || segid_sh[s + 1] != id)
            seg_count[b * S + id] = s + 1 - seg_start[b * S + id];
    }
    // short mask output
    int cnt = cnt_sh;
    for (int s = tid; s < S; s += 256)
        out[SMASK_OFF + b * S + s] = (s < cnt) ? 1.f : 0.f;
}

// ============================================================
// Kernel 4: scalar loss / num_b / total
// ============================================================
__global__ void finalize_kernel(const float* __restrict__ row_stats, float* __restrict__ out) {
    float num_b = 0.f, total = 0.f;
    for (int b = 0; b < BS; ++b) { num_b += row_stats[b * 2]; total += row_stats[b * 2 + 1]; }
    float lp = lgammaf(total + 1.f) - lgammaf(num_b + 1.f) - lgammaf(total - num_b + 1.f)
             + num_b * logf(PRIOR) + (total - num_b) * logf(1.f - PRIOR);
    out[LOSS_OFF] = -lp / total;
    out[NUMB_OFF] = num_b;
    out[TOT_OFF]  = total;
}

// ============================================================
// Kernel 5: segment mean-pool (contiguous ranges, coalesced)
// ============================================================
__global__ __launch_bounds__(256)
void pool_kernel(const float* __restrict__ hidden,
                 const int* __restrict__ seg_start,
                 const int* __restrict__ seg_count,
                 float* __restrict__ out_pooled) {
    const int b = blockIdx.y;
    const int s = blockIdx.x;
    const int tid = threadIdx.x;
    const int n  = seg_count[b * S + s];
    const int st = seg_start[b * S + s];
    const float denom = (float)n + EPS;

    float acc0 = 0.f, acc1 = 0.f, acc2 = 0.f, acc3 = 0.f;
    const float* base = hidden + ((size_t)b * S + st) * D;
    for (int t = 0; t < n; ++t) {
        const float* row = base + (size_t)t * D;
        acc0 += row[tid];
        acc1 += row[tid + 256];
        acc2 += row[tid + 512];
        acc3 += row[tid + 768];
    }
    float* o = out_pooled + ((size_t)b * S + s) * D;
    o[tid]       = acc0 / denom;   // n==0 -> 0/1e-9 = 0, matches reference
    o[tid + 256] = acc1 / denom;
    o[tid + 512] = acc2 / denom;
    o[tid + 768] = acc3 / denom;
}

// ============================================================
extern "C" void kernel_launch(void* const* d_in, const int* in_sizes, int n_in,
                              void* d_out, int out_size, void* d_ws, size_t ws_size,
                              hipStream_t stream) {
    const float* hidden = (const float*)d_in[0];
    const float* amask  = (const float*)d_in[1];
    const float* noise  = (const float*)d_in[2];
    const float* W1     = (const float*)d_in[3];
    const float* b1     = (const float*)d_in[4];
    const float* W2     = (const float*)d_in[5];
    const float* b2     = (const float*)d_in[6];
    float* out = (float*)d_out;

    // workspace carve-up
    char* ws = (char*)d_ws;
    float* partial   = (float*)ws;                                   // NBLK*TOK f32 (2 MB)
    float* logits    = (float*)(ws + (size_t)NBLK * TOK * 4);        // TOK f32
    int*   seg_start = (int*)  (ws + (size_t)NBLK * TOK * 4 + TOK * 4);
    int*   seg_count = seg_start + BS * S;
    float* row_stats = (float*)(seg_count + BS * S);

    gate_gemm_kernel<<<dim3(NBLK, MBLK), 256, 0, stream>>>(hidden, W1, b1, W2, partial);
    reduce_logits_kernel<<<(TOK + 255) / 256, 256, 0, stream>>>(partial, logits);
    boundary_kernel<<<BS, 256, 0, stream>>>(logits, amask, noise, b2,
                                            seg_start, seg_count, row_stats, out);
    finalize_kernel<<<1, 1, 0, stream>>>(row_stats, out);
    pool_kernel<<<dim3(S, BS), 256, 0, stream>>>(hidden, seg_start, seg_count, out);
}
